// DeltaNet_31877247271465
// MI455X (gfx1250) — matrix operand order repaired and checked by reading, not compiled, not run on hardware
//
#include <hip/hip_runtime.h>
#include <math.h>

// ---------------- problem constants (match reference) ----------------
#define BB   2
#define LL   4096
#define DD   1024
#define HH   4
#define DKH  256          // head dim (k and v)
#define CHK  32           // chunk
#define NCH  (LL / CHK)   // 128 chunks
#define GHID 1024         // gate hidden
#define KPAD 1056         // 1040 padded to multiple of 32
#define SSTR 264          // LDS row stride (bf16) for 256-wide tiles; 528B rows, 16B aligned
#define TSTR 40           // LDS row stride (bf16) for 32-wide transposed tiles; 80B rows
#define NTOK (BB * LL)            // 8192 tokens
#define NTH  (BB * LL * HH)       // 32768 token-heads
#define NEL  ((size_t)BB * LL * DD)  // 8388608 activation elements

typedef __attribute__((ext_vector_type(16))) __bf16 bf16x16;
typedef __attribute__((ext_vector_type(8)))  __bf16 bf16x8;
typedef __attribute__((ext_vector_type(8)))  float  v8f;

// ---------------- WMMA helpers (CDNA5 16x16x32 bf16, fp32 accum) -----
__device__ inline v8f wmma_bf(bf16x16 a, bf16x16 b, v8f c) {
  // v_wmma_f32_16x16x32_bf16  (neg_a, A, neg_b, B, c_mod, C, reuse_a, reuse_b)
  return __builtin_amdgcn_wmma_f32_16x16x32_bf16(false, a, false, b, (short)0, c,
                                                 false, false);
}
__device__ inline v8f zero8() {
  v8f z;
  for (int r = 0; r < 8; ++r) z[r] = 0.f;
  return z;
}
__device__ inline bf16x16 combine16(bf16x8 lo, bf16x8 hi) {
  bf16x16 r;
#pragma unroll
  for (int i = 0; i < 8; ++i) { r[i] = lo[i]; r[8 + i] = hi[i]; }
  return r;
}

// A fragment: 16x32 (MxK) bf16, row-major source (ld in elements).
// ISA 7.12.2 layout: lane m=lane&15; lanes 16-31 take K offset +8.
// Element order: a[0..7] = A[m][koff+0..7], a[8..15] = A[m][koff+16..23]
// -> two aligned 16B vector loads per lane (ds_load_b128).
__device__ inline bf16x16 frag_a_pk(const __bf16* base, int m0, int k0, int lane, int ld) {
  int m = m0 + (lane & 15);
  int koff = k0 + ((lane >> 4) << 3);
  const __bf16* p = base + (size_t)m * ld + koff;
  bf16x8 lo = *(const bf16x8*)p;
  bf16x8 hi = *(const bf16x8*)(p + 16);
  return combine16(lo, hi);
}
// B fragment: 32x16 (KxN) where the source is stored K-contiguous per N row:
// srcT[n][k] = B[k][n].  b[0..7] = B[kh+0..7][n], b[8..15] = B[kh+16..23][n] (kh = k0 + 8*(lane>>4)).
__device__ inline bf16x16 frag_b_pk(const __bf16* srcT, int k0, int n0, int lane, int ld) {
  int n = n0 + (lane & 15);
  int kh = k0 + ((lane >> 4) << 3);
  const __bf16* p = srcT + (size_t)n * ld + kh;
  bf16x8 lo = *(const bf16x8*)p;
  bf16x8 hi = *(const bf16x8*)(p + 16);
  return combine16(lo, hi);
}

// ---------------- small utility kernels ------------------------------
__global__ __launch_bounds__(256) void cvt_f32_bf16_kernel(const float* __restrict__ src,
                                                           __bf16* __restrict__ dst, int n) {
  int i = blockIdx.x * 256 + threadIdx.x;
  if (i < n) dst[i] = (__bf16)src[i];
}

__global__ __launch_bounds__(256) void padw1_kernel(const float* __restrict__ w1,
                                                    __bf16* __restrict__ dst) {
  int i = blockIdx.x * 256 + threadIdx.x;  // over KPAD*GHID
  if (i >= KPAD * GHID) return;
  int r = i >> 10, c = i & 1023;
  dst[i] = (__bf16)((r < DD + 16) ? w1[r * GHID + c] : 0.f);
}

// beta = sigmoid(hs @ Wb), one thread per (token, head)
__global__ __launch_bounds__(256) void beta_kernel(const float* __restrict__ hs,
                                                   const float* __restrict__ Wb,
                                                   float* __restrict__ beta) {
  int i = blockIdx.x * 256 + threadIdx.x;  // NTH
  if (i >= NTH) return;
  int h = i & (HH - 1);
  size_t bl = (size_t)(i >> 2);
  const float* row = hs + bl * DD;
  float acc = 0.f;
  for (int d = 0; d < DD; ++d) acc += row[d] * Wb[d * HH + h];
  beta[i] = 1.f / (1.f + expf(-acc));
}

// causal depthwise conv (torch cross-correlation), optional SiLU
__global__ __launch_bounds__(256) void dwconv_kernel(const __bf16* __restrict__ src,
                                                     const float* __restrict__ w,
                                                     __bf16* __restrict__ dst,
                                                     int taps, int act_silu) {
  size_t i = (size_t)blockIdx.x * 256 + threadIdx.x;  // over NEL
  if (i >= NEL) return;
  int c = (int)(i % DD);
  size_t bl = i / DD;
  int l = (int)(bl % LL);
  size_t b = bl / LL;
  float acc = 0.f;
  for (int t = 0; t < taps; ++t) {
    int ls = l - (taps - 1) + t;
    if (ls >= 0) acc += (float)src[(b * LL + ls) * DD + c] * w[c * taps + t];
  }
  if (act_silu) acc = acc / (1.f + expf(-acc));  // x*sigmoid(x)
  dst[i] = (__bf16)acc;
}

// ---------------- generic WMMA GEMM ---------------------------------
// C[M,N] = act(A[M,K] @ B[K,N] + bias); A,B bf16 row-major (lda=K, ldb=N).
// mode: 0 -> store fp32, 1 -> store bf16, 2 -> GELU(exact) -> bf16
// Block 256 thr (8 waves), tile 64(M) x 128(N), BK=32; wave = 32x32 (2x2 tiles).
// B tile is staged TRANSPOSED in LDS so B fragments are 2x ds_load_b128 per lane.
__global__ __launch_bounds__(256) void gemm_bf16_wmma(const __bf16* __restrict__ A,
                                                      const __bf16* __restrict__ Bm,
                                                      void* __restrict__ Cout,
                                                      int M, int N, int K,
                                                      const float* __restrict__ bias,
                                                      int mode) {
  __shared__ __bf16 As[64][40];    // 64 rows x 32 K (+pad)
  __shared__ __bf16 Bt[128][40];   // transposed: Bt[n][k], 128 N x 32 K (+pad)
  int tid = threadIdx.x, lane = tid & 31, wid = tid >> 5;
  int bm = blockIdx.y * 64, bn = blockIdx.x * 128;
  int wm = (wid >> 2) * 32, wn = (wid & 3) * 32;

  v8f acc[2][2];
  for (int i = 0; i < 2; ++i)
    for (int j = 0; j < 2; ++j) acc[i][j] = zero8();

  int ar = tid >> 2, ac = (tid & 3) * 8;   // 64 rows x 32 cols, 8 per thread
  int br = tid >> 3, bc = (tid & 7) * 16;  // 32 K-rows x 128 N, 16 per thread

  for (int k0 = 0; k0 < K; k0 += 32) {
#pragma unroll
    for (int j = 0; j < 8; ++j)
      As[ar][ac + j] = A[(size_t)(bm + ar) * K + k0 + ac + j];
#pragma unroll
    for (int j = 0; j < 16; ++j)  // coalesced global read, transposed LDS write
      Bt[bc + j][br] = Bm[(size_t)(k0 + br) * N + bn + bc + j];
    if (k0 + 32 < K)  // global_prefetch_b8 for next A tile
      __builtin_prefetch(&A[(size_t)(bm + ar) * K + k0 + 32 + ac], 0, 0);
    __syncthreads();
#pragma unroll
    for (int i = 0; i < 2; ++i) {
      bf16x16 af = frag_a_pk(&As[0][0], wm + 16 * i, 0, lane, 40);
#pragma unroll
      for (int j = 0; j < 2; ++j) {
        bf16x16 bf = frag_b_pk(&Bt[0][0], 0, wn + 16 * j, lane, 40);
        acc[i][j] = wmma_bf(af, bf, acc[i][j]);
      }
    }
    __syncthreads();
  }
  // epilogue: C layout -> VGPR r holds M=r (lanes 0-15) / M=r+8 (lanes 16-31)
#pragma unroll
  for (int i = 0; i < 2; ++i)
#pragma unroll
    for (int j = 0; j < 2; ++j)
#pragma unroll
      for (int r = 0; r < 8; ++r) {
        int m = bm + wm + 16 * i + r + ((lane >> 4) << 3);
        int n = bn + wn + 16 * j + (lane & 15);
        float val = acc[i][j][r];
        if (bias) val += bias[n];
        if (mode == 2) val = 0.5f * val * (1.f + erff(val * 0.70710678118f));
        if (mode == 0)
          ((float*)Cout)[(size_t)m * N + n] = val;
        else
          ((__bf16*)Cout)[(size_t)m * N + n] = (__bf16)val;
      }
}

// ---------------- delta-rule chunkwise kernel ------------------------
// One block per (b,h); transposed state ST[dv][dk] (256x256 bf16) resident in
// LDS for all 128 chunks.  All WMMA operands are packed-contiguous per lane.
__global__ __launch_bounds__(256) void delta_kernel(const __bf16* __restrict__ qg,
                                                    const __bf16* __restrict__ kg,
                                                    const __bf16* __restrict__ vg,
                                                    const float* __restrict__ betag,
                                                    __bf16* __restrict__ og) {
  extern __shared__ char smem_raw[];
  __bf16* ST  = (__bf16*)smem_raw;           // 256 x SSTR : ST[n=dv][k=dk] = S[k][n]
  __bf16* qn  = ST  + 256 * SSTR;            // 32 x SSTR  (row-major, A/Bt operand)
  __bf16* kn  = qn  + 32 * SSTR;
  __bf16* kb  = kn  + 32 * SSTR;
  __bf16* uu  = kb  + 32 * SSTR;             // raw v (ph1-2), then u (ph5+)
  __bf16* ww  = uu  + 32 * SSTR;
  __bf16* knT = ww  + 32 * SSTR;             // 256 x TSTR : knT[d][c]
  __bf16* kbT = knT + 256 * TSTR;
  __bf16* vbT = kbT + 256 * TSTR;
  __bf16* uiT = vbT + 256 * TSTR;
  float*  Am  = (float*)(uiT + 256 * TSTR);  // 32 x 33 fp32 (triangular solve)
  float*  Tm  = Am + 32 * 33;                // 32 x 33 fp32
  __bf16* Atb = (__bf16*)(Tm + 32 * 33);     // attn bf16, 32 x TSTR
  __bf16* Tb  = Atb + 32 * TSTR;             // T bf16, 32 x TSTR
  float*  red = (float*)(Tb + 32 * TSTR);    // 64 row sums

  int tid = threadIdx.x, lane = tid & 31, wid = tid >> 5;
  int b = blockIdx.x >> 2, h = blockIdx.x & 3;

  for (int i = tid; i < 256 * SSTR; i += 256) ST[i] = (__bf16)0.f;
  __syncthreads();

  for (int t = 0; t < NCH; ++t) {
    int l0 = t * CHK;
    if (tid < 64) red[tid] = 0.f;
    __syncthreads();

    // phase 1: stage raw q,k,v ; row sum-of-squares for l2norm
    {
      int row = tid >> 3, c0 = (tid & 7) * 32;
      size_t gbase = ((size_t)(b * LL + l0 + row)) * DD + h * DKH + c0;
      float qss = 0.f, kss = 0.f;
      for (int j = 0; j < 32; ++j) {
        float qv = (float)qg[gbase + j];
        float kv = (float)kg[gbase + j];
        float vv = (float)vg[gbase + j];
        qn[row * SSTR + c0 + j] = (__bf16)qv;
        kn[row * SSTR + c0 + j] = (__bf16)kv;
        uu[row * SSTR + c0 + j] = (__bf16)vv;
        qss += qv * qv;
        kss += kv * kv;
      }
      atomicAdd(&red[row], qss);       // ds_add_f32
      atomicAdd(&red[32 + row], kss);
    }
    __syncthreads();

    // phase 2: normalize q,k; build kb and transposed copies knT/kbT/vbT
    {
      int row = tid >> 3, c0 = (tid & 7) * 32;
      float qs = rsqrtf(red[row] + 1e-12f);
      float ks = rsqrtf(red[32 + row] + 1e-12f);
      float bv = betag[((size_t)(b * LL + l0 + row)) * HH + h];
      for (int j = 0; j < 32; ++j) {
        int d = c0 + j;
        int idx = row * SSTR + d;
        float qv = (float)qn[idx] * qs;
        float kv = (float)kn[idx] * ks;
        float kbv = kv * bv;
        float vbv = (float)uu[idx] * bv;
        qn[idx] = (__bf16)qv;
        kn[idx] = (__bf16)kv;
        kb[idx] = (__bf16)kbv;
        knT[d * TSTR + row] = (__bf16)kv;
        kbT[d * TSTR + row] = (__bf16)kbv;
        vbT[d * TSTR + row] = (__bf16)vbv;
      }
    }
    __syncthreads();

    // phase 3: A = strict_tril(kb @ kn^T) -> fp32 ; attn = tril(qn @ kn^T) -> bf16
    {
      int isAttn = wid >> 2;
      int tt = wid & 3;
      int m0 = (tt >> 1) * 16, n0 = (tt & 1) * 16;
      const __bf16* asrc = isAttn ? qn : kb;
      v8f acc = zero8();
      for (int kk = 0; kk < DKH; kk += 32) {
        bf16x16 af = frag_a_pk(asrc, m0, kk, lane, SSTR);
        bf16x16 bf = frag_b_pk(kn, kk, n0, lane, SSTR);  // kn rows are the N axis
        acc = wmma_bf(af, bf, acc);
      }
#pragma unroll
      for (int r = 0; r < 8; ++r) {
        int m = m0 + r + ((lane >> 4) << 3);
        int n = n0 + (lane & 15);
        if (isAttn)
          Atb[m * TSTR + n] = (__bf16)((m >= n) ? acc[r] : 0.f);
        else
          Am[m * 33 + n] = (m > n) ? acc[r] : 0.f;
      }
    }
    __syncthreads();

    // phase 4: T = (I + A)^-1 forward substitution (one wave; lane = column)
    if (wid == 0) {
      for (int i = 0; i < 32; ++i) Tm[i * 33 + lane] = (i == lane) ? 1.f : 0.f;
      for (int i = 1; i < 32; ++i) {
        float s = (i == lane) ? 1.f : 0.f;
        for (int p = 0; p < i; ++p) s -= Am[i * 33 + p] * Tm[p * 33 + lane];
        Tm[i * 33 + lane] = s;
      }
    }
    __syncthreads();
    for (int i = tid; i < 1024; i += 256)
      Tb[(i >> 5) * TSTR + (i & 31)] = (__bf16)Tm[(i >> 5) * 33 + (i & 31)];
    __syncthreads();

    // phase 5: u = T @ vb, w = T @ kb   (64 tiles, K=32 -> 1 WMMA each)
    for (int q = 0; q < 8; ++q) {
      int tt = wid * 8 + q;
      int which = tt >> 5;       // 0: u, 1: w
      int local = tt & 31;
      int m0 = (local >> 4) * 16, n0 = (local & 15) * 16;
      const __bf16* bsrcT = which ? kbT : vbT;
      __bf16* dst = which ? ww : uu;
      bf16x16 af = frag_a_pk(Tb, m0, 0, lane, TSTR);
      bf16x16 bf = frag_b_pk(bsrcT, 0, n0, lane, TSTR);
      v8f acc = wmma_bf(af, bf, zero8());
#pragma unroll
      for (int r = 0; r < 8; ++r) {
        int m = m0 + r + ((lane >> 4) << 3);
        int n = n0 + (lane & 15);
        dst[m * SSTR + n] = (__bf16)acc[r];
      }
    }
    __syncthreads();

    // phase 6: ui = u - w @ S   (32 tiles, K=256); store transposed uiT[d][c]
    for (int q = 0; q < 4; ++q) {
      int tt = wid * 4 + q;
      int m0 = (tt >> 4) * 16, n0 = (tt & 15) * 16;
      v8f acc = zero8();
      for (int kk = 0; kk < DKH; kk += 32) {
        bf16x16 af = frag_a_pk(ww, m0, kk, lane, SSTR);
        bf16x16 bf = frag_b_pk(ST, kk, n0, lane, SSTR);  // ST[n][k] = S[k][n]
        acc = wmma_bf(af, bf, acc);
      }
#pragma unroll
      for (int r = 0; r < 8; ++r) {
        int m = m0 + r + ((lane >> 4) << 3);
        int n = n0 + (lane & 15);
        uiT[n * TSTR + m] = (__bf16)((float)uu[m * SSTR + n] - acc[r]);
      }
    }
    __syncthreads();

    // phase 7: o = qn @ S + attn @ ui ; write to global
    for (int q = 0; q < 4; ++q) {
      int tt = wid * 4 + q;
      int m0 = (tt >> 4) * 16, n0 = (tt & 15) * 16;
      v8f acc = zero8();
      for (int kk = 0; kk < DKH; kk += 32) {
        bf16x16 af = frag_a_pk(qn, m0, kk, lane, SSTR);
        bf16x16 bf = frag_b_pk(ST, kk, n0, lane, SSTR);
        acc = wmma_bf(af, bf, acc);
      }
      bf16x16 af2 = frag_a_pk(Atb, m0, 0, lane, TSTR);
      bf16x16 bf2 = frag_b_pk(uiT, 0, n0, lane, TSTR);
      acc = wmma_bf(af2, bf2, acc);
#pragma unroll
      for (int r = 0; r < 8; ++r) {
        int m = m0 + r + ((lane >> 4) << 3);
        int n = n0 + (lane & 15);
        og[((size_t)(b * LL + l0 + m)) * DD + h * DKH + n] = (__bf16)acc[r];
      }
    }
    __syncthreads();

    // phase 8: S += kn^T @ ui  (256 tiles, K=32); C/D live in ST transposed
    for (int q = 0; q < 32; ++q) {
      int tt = wid * 32 + q;
      int m0 = (tt >> 4) * 16, n0 = (tt & 15) * 16;
      bf16x16 af = frag_a_pk(knT, m0, 0, lane, TSTR);
      bf16x16 bf = frag_b_pk(uiT, 0, n0, lane, TSTR);
      v8f acc;
#pragma unroll
      for (int r = 0; r < 8; ++r) {
        int m = m0 + r + ((lane >> 4) << 3);
        int n = n0 + (lane & 15);
        acc[r] = (float)ST[n * SSTR + m];
      }
      acc = wmma_bf(af, bf, acc);
#pragma unroll
      for (int r = 0; r < 8; ++r) {
        int m = m0 + r + ((lane >> 4) << 3);
        int n = n0 + (lane & 15);
        ST[n * SSTR + m] = (__bf16)acc[r];
      }
    }
    __syncthreads();
  }
}

// ---------------- stats + gate-input assembly ------------------------
// one wave (32 lanes) per (b,l,h); gate_in row = [hs(1024), 16 stats, 16 zero-pad]
__global__ __launch_bounds__(256) void stats_gate_kernel(const __bf16* __restrict__ hs_bf,
                                                         const __bf16* __restrict__ ls,
                                                         const __bf16* __restrict__ llb,
                                                         const __bf16* __restrict__ dl,
                                                         const __bf16* __restrict__ vv,
                                                         __bf16* __restrict__ gate_in) {
  int wid = threadIdx.x >> 5, lane = threadIdx.x & 31;
  int gw = blockIdx.x * 8 + wid;
  if (gw >= NTH) return;
  int h = gw & 3;
  size_t bl = (size_t)(gw >> 2);
  size_t rb = bl * DD + h * DKH;
  __bf16* gi = gate_in + (size_t)gw * KPAD;
  const __bf16* srcs[4] = {ls + rb, llb + rb, dl + rb, vv + rb};
  for (int s = 0; s < 4; ++s) {
    float sum = 0.f, ss = 0.f, sa = 0.f;
    for (int j = lane; j < DKH; j += 32) {
      float x = (float)srcs[s][j];
      sum += x; ss += x * x; sa += fabsf(x);
    }
    for (int m = 16; m >= 1; m >>= 1) {
      sum += __shfl_xor(sum, m, 32);
      ss  += __shfl_xor(ss, m, 32);
      sa  += __shfl_xor(sa, m, 32);
    }
    if (lane == 0) {
      float mean = sum * (1.f / 256.f);
      gi[1024 + s * 4 + 0] = (__bf16)mean;
      gi[1024 + s * 4 + 1] = (__bf16)(ss * (1.f / 256.f) - mean * mean);
      gi[1024 + s * 4 + 2] = (__bf16)(sa * (1.f / 256.f));
      gi[1024 + s * 4 + 3] = (__bf16)sqrtf(ss);
    }
  }
  const __bf16* hrow = hs_bf + bl * DD;
  for (int j = lane; j < DD; j += 32) gi[j] = hrow[j];
  if (lane < 16) gi[1040 + lane] = (__bf16)0.f;
}

// ---------------- final fusion: logits/softmax/mix/RMSNorm -----------
__global__ __launch_bounds__(256) void final_kernel(const __bf16* __restrict__ hgate,
                                                    const float* __restrict__ w2,
                                                    const float* __restrict__ b2,
                                                    const float* __restrict__ logit_temp,
                                                    const float* __restrict__ conv_res_logit,
                                                    const float* __restrict__ rms_w,
                                                    const __bf16* __restrict__ ls,
                                                    const __bf16* __restrict__ llb,
                                                    const __bf16* __restrict__ dl,
                                                    const __bf16* __restrict__ vv,
                                                    __bf16* __restrict__ opre) {
  int wid = threadIdx.x >> 5, lane = threadIdx.x & 31;
  int gw = blockIdx.x * 8 + wid;
  if (gw >= NTH) return;
  int h = gw & 3;
  size_t bl = (size_t)(gw >> 2);
  const __bf16* hg = hgate + (size_t)gw * GHID;
  float a0 = 0.f, a1 = 0.f, a2 = 0.f, a3 = 0.f;
  for (int j = lane; j < GHID; j += 32) {
    float x = (float)hg[j];
    a0 += x * w2[j * 4 + 0];
    a1 += x * w2[j * 4 + 1];
    a2 += x * w2[j * 4 + 2];
    a3 += x * w2[j * 4 + 3];
  }
  for (int m = 16; m >= 1; m >>= 1) {
    a0 += __shfl_xor(a0, m, 32);
    a1 += __shfl_xor(a1, m, 32);
    a2 += __shfl_xor(a2, m, 32);
    a3 += __shfl_xor(a3, m, 32);
  }
  float temp = log1pf(expf(logit_temp[0])) + 1e-6f;  // softplus
  float lg[4] = {(a0 + b2[0]) / temp, (a1 + b2[1]) / temp,
                 (a2 + b2[2]) / temp, (a3 + b2[3]) / temp};
  float mx = fmaxf(fmaxf(lg[0], lg[1]), fmaxf(lg[2], lg[3]));
  float e0 = expf(lg[0] - mx), e1 = expf(lg[1] - mx), e2 = expf(lg[2] - mx),
        e3 = expf(lg[3] - mx);
  float inv = 1.f / (e0 + e1 + e2 + e3);
  const float keep = 1.f - 2.f * 0.08f;
  float p0 = e0 * inv * keep + 0.08f;
  float p1 = e1 * inv * keep + 0.08f;
  float p2 = e2 * inv * keep;
  float p3 = e3 * inv * keep;
  float resid = 1.f / (1.f + expf(-conv_res_logit[h]));
  size_t rb = bl * DD + h * DKH;
  float ov[8];
  float ss = 0.f;
#pragma unroll
  for (int j = 0; j < 8; ++j) {
    int d = lane + 32 * j;
    float fs = (float)ls[rb + d];
    float o = p0 * fs + p1 * (float)llb[rb + d] + p2 * (float)dl[rb + d] +
              p3 * (float)vv[rb + d] + resid * fs;
    ov[j] = o;
    ss += o * o;
  }
  for (int m = 16; m >= 1; m >>= 1) ss += __shfl_xor(ss, m, 32);
  float rn = rsqrtf(ss * (1.f / 256.f) + 1e-5f);
#pragma unroll
  for (int j = 0; j < 8; ++j) {
    int d = lane + 32 * j;
    opre[rb + d] = (__bf16)(ov[j] * rn * rms_w[d]);
  }
}

// ---------------- host orchestration ---------------------------------
extern "C" void kernel_launch(void* const* d_in, const int* in_sizes, int n_in,
                              void* d_out, int out_size, void* d_ws, size_t ws_size,
                              hipStream_t stream) {
  const float* hs      = (const float*)d_in[0];
  const float* Wq      = (const float*)d_in[1];
  const float* Wk      = (const float*)d_in[2];
  const float* Wv      = (const float*)d_in[3];
  const float* Wb      = (const float*)d_in[4];
  const float* cwq     = (const float*)d_in[5];
  const float* cwk     = (const float*)d_in[6];
  const float* cwv     = (const float*)d_in[7];
  const float* fir_s   = (const float*)d_in[8];
  const float* fir_l   = (const float*)d_in[9];
  const float* w1      = (const float*)d_in[10];
  const float* b1      = (const float*)d_in[11];
  const float* w2      = (const float*)d_in[12];
  const float* b2      = (const float*)d_in[13];
  const float* ltemp   = (const float*)d_in[14];
  const float* creslog = (const float*)d_in[15];
  const float* rms_w   = (const float*)d_in[16];
  const float* Wo      = (const float*)d_in[17];
  float* out = (float*)d_out;

  char* ws = (char*)d_ws;
  size_t off = 0;
  auto alloc = [&](size_t bytes) -> char* {
    char* p = ws + off;
    off = (off + bytes + 255) & ~(size_t)255;
    return p;
  };
  __bf16* hs_bf   = (__bf16*)alloc(NEL * 2);
  __bf16* Wq_bf   = (__bf16*)alloc((size_t)DD * DD * 2);
  __bf16* Wk_bf   = (__bf16*)alloc((size_t)DD * DD * 2);
  __bf16* Wv_bf   = (__bf16*)alloc((size_t)DD * DD * 2);
  __bf16* Wo_bf   = (__bf16*)alloc((size_t)DD * DD * 2);
  __bf16* w1_bf   = (__bf16*)alloc((size_t)KPAD * GHID * 2);
  __bf16* qlin    = (__bf16*)alloc(NEL * 2);
  __bf16* klin    = (__bf16*)alloc(NEL * 2);
  __bf16* vlin    = (__bf16*)alloc(NEL * 2);
  __bf16* qc      = (__bf16*)alloc(NEL * 2);
  __bf16* kc      = (__bf16*)alloc(NEL * 2);
  __bf16* vc      = (__bf16*)alloc(NEL * 2);
  float*  beta_f  = (float*)alloc((size_t)NTH * 4);
  __bf16* ls_bf   = (__bf16*)alloc(NEL * 2);
  __bf16* llb_bf  = (__bf16*)alloc(NEL * 2);
  __bf16* delta   = (__bf16*)alloc(NEL * 2);
  __bf16* gate_in = (__bf16*)alloc((size_t)NTH * KPAD * 2);
  __bf16* hgate   = (__bf16*)alloc((size_t)NTH * GHID * 2);
  __bf16* opre    = (__bf16*)alloc(NEL * 2);
  (void)ws_size; (void)n_in; (void)in_sizes; (void)out_size;

  // 1) fp32 -> bf16 conversions
  cvt_f32_bf16_kernel<<<(int)(NEL / 256), 256, 0, stream>>>(hs, hs_bf, (int)NEL);
  cvt_f32_bf16_kernel<<<DD * DD / 256, 256, 0, stream>>>(Wq, Wq_bf, DD * DD);
  cvt_f32_bf16_kernel<<<DD * DD / 256, 256, 0, stream>>>(Wk, Wk_bf, DD * DD);
  cvt_f32_bf16_kernel<<<DD * DD / 256, 256, 0, stream>>>(Wv, Wv_bf, DD * DD);
  cvt_f32_bf16_kernel<<<DD * DD / 256, 256, 0, stream>>>(Wo, Wo_bf, DD * DD);
  padw1_kernel<<<(KPAD * GHID + 255) / 256, 256, 0, stream>>>(w1, w1_bf);

  // 2) q/k/v projections (WMMA bf16), 8192x1024x1024 each
  dim3 gproj(DD / 128, NTOK / 64);
  gemm_bf16_wmma<<<gproj, 256, 0, stream>>>(hs_bf, Wq_bf, qlin, NTOK, DD, DD, nullptr, 1);
  gemm_bf16_wmma<<<gproj, 256, 0, stream>>>(hs_bf, Wk_bf, klin, NTOK, DD, DD, nullptr, 1);
  gemm_bf16_wmma<<<gproj, 256, 0, stream>>>(hs_bf, Wv_bf, vlin, NTOK, DD, DD, nullptr, 1);

  // 3) beta gate
  beta_kernel<<<NTH / 256, 256, 0, stream>>>(hs, Wb, beta_f);

  // 4) causal depthwise conv + SiLU
  int nblk = (int)(NEL / 256);
  dwconv_kernel<<<nblk, 256, 0, stream>>>(qlin, cwq, qc, 4, 1);
  dwconv_kernel<<<nblk, 256, 0, stream>>>(klin, cwk, kc, 4, 1);
  dwconv_kernel<<<nblk, 256, 0, stream>>>(vlin, cwv, vc, 4, 1);

  // 5) FIR branches on v
  dwconv_kernel<<<nblk, 256, 0, stream>>>(vc, fir_s, ls_bf, 5, 0);
  dwconv_kernel<<<nblk, 256, 0, stream>>>(vc, fir_l, llb_bf, 64, 0);

  // 6) delta-rule chunkwise recurrence: 8 workgroups, state in LDS
  size_t dsmem = (size_t)(256 * SSTR + 5 * 32 * SSTR + 4 * 256 * TSTR) * 2 +
                 (size_t)2 * 32 * 33 * 4 + (size_t)2 * 32 * TSTR * 2 + 64 * 4;
  delta_kernel<<<BB * HH, 256, dsmem, stream>>>(qc, kc, vc, beta_f, delta);

  // 7) stats + gate input assembly
  stats_gate_kernel<<<NTH / 8, 256, 0, stream>>>(hs_bf, ls_bf, llb_bf, delta, vc, gate_in);

  // 8) gate MLP layer 1 (WMMA): 32768 x 1024 x 1056, bias + exact GELU
  dim3 ggate(GHID / 128, NTH / 64);
  gemm_bf16_wmma<<<ggate, 256, 0, stream>>>(gate_in, w1_bf, hgate, NTH, GHID, KPAD, b1, 2);

  // 9) gate logits + softmax + floor + residual + RMSNorm
  final_kernel<<<NTH / 8, 256, 0, stream>>>(hgate, w2, b2, ltemp, creslog, rms_w,
                                            ls_bf, llb_bf, delta, vc, opre);

  // 10) output projection -> fp32 d_out
  gemm_bf16_wmma<<<gproj, 256, 0, stream>>>(opre, Wo_bf, out, NTOK, DD, DD, nullptr, 0);
}
